// MultiheadSelfAttention_2491081032144
// MI455X (gfx1250) — matrix operand order repaired and checked by reading, not compiled, not run on hardware
//
#include <hip/hip_runtime.h>
#include <stdint.h>

#define DM 1024
#define NH 16
#define DH 64
#define SS 2048
#define BB 2

typedef __attribute__((ext_vector_type(16))) __bf16 v16bf;
typedef __attribute__((ext_vector_type(8)))  __bf16 v8bf;
typedef __attribute__((ext_vector_type(8)))  float  v8f;

__device__ inline __bf16 f2bf(float f) {
  unsigned u = __builtin_bit_cast(unsigned, f);
  unsigned r = (u + 0x7FFFu + ((u >> 16) & 1u)) >> 16;
  unsigned short s = (unsigned short)r;
  return __builtin_bit_cast(__bf16, s);
}

__device__ inline v16bf mk16(v8bf lo, v8bf hi) {
  v16bf r;
#pragma unroll
  for (int i = 0; i < 8; ++i) { r[i] = lo[i]; r[i + 8] = hi[i]; }
  return r;
}
// A-fragment: two disjoint 8-element chunks (K 16*sel..+7 / 16*sel+8..+15 per lane half)
__device__ inline v16bf ld_pair(const __bf16* p0, const __bf16* p1) {
  return mk16(*(const v8bf*)p0, *(const v8bf*)p1);
}
// B-fragment: 16 contiguous elements (K 16*sel..+15 at column = lane&15)
__device__ inline v16bf ld_c16(const __bf16* p) {
  return mk16(*(const v8bf*)p, *(const v8bf*)(p + 8));
}
__device__ inline v8f zero8() {
  v8f z;
#pragma unroll
  for (int i = 0; i < 8; ++i) z[i] = 0.f;
  return z;
}
__device__ inline float rmax16(float v) {
#pragma unroll
  for (int m = 1; m <= 8; m <<= 1) v = fmaxf(v, __shfl_xor(v, m, 32));
  return v;
}
__device__ inline float rsum16(float v) {
#pragma unroll
  for (int m = 1; m <= 8; m <<= 1) v += __shfl_xor(v, m, 32);
  return v;
}
__device__ inline v8f wmma_bf16(v16bf a, v16bf b, v8f c) {
  return __builtin_amdgcn_wmma_f32_16x16x32_bf16(false, a, false, b, (short)0, c,
                                                 false, false);
}

// ---------------------------------------------------------------- convert
__global__ void f32_to_bf16(const float* __restrict__ src, __bf16* __restrict__ dst,
                            int n) {
  for (int i = blockIdx.x * blockDim.x + threadIdx.x; i < n;
       i += gridDim.x * blockDim.x)
    dst[i] = f2bf(src[i]);
}

// ---------------------------------------------------------------- QKV GEMM
// 64x64 output tile per wave (4x4 WMMA accumulators): each A fragment reused
// across 4 N-subtiles, each B fragment across 4 M-subtiles -> 1 load : 1 wmma.
// __launch_bounds__(256,1): allow full 256-VGPR allocation (acc=128 + frags=64)
// so nothing spills to scratch.
// out[s,e] = sum_d X[s,d] * Wqkv[e,d]; the 64-wide N block is exactly one
// head: write Q,K as [b,h,s,64] (Q pre-scaled by 1/8), V^T as [b,h,64,s].
__global__ __launch_bounds__(256, 1) void qkv_gemm(
    const __bf16* __restrict__ X, const __bf16* __restrict__ W,
    __bf16* __restrict__ Q, __bf16* __restrict__ K, __bf16* __restrict__ Vt) {
  const int wave = threadIdx.x >> 5, lane = threadIdx.x & 31;
  const int lh = lane & 15, sel = lane >> 4;
  const int NT = (3 * DM) / 64;  // 48 column blocks
  const int tile = blockIdx.x * 8 + wave;
  const int m0 = (tile / NT) * 64;
  const int e0 = (tile % NT) * 64;
  const __bf16* a0 = X + (size_t)(m0 + lh) * DM;
  const __bf16* b0 = W + (size_t)(e0 + lh) * DM;

  v8f acc[4][4];
#pragma unroll
  for (int mi = 0; mi < 4; ++mi)
#pragma unroll
    for (int ni = 0; ni < 4; ++ni) acc[mi][ni] = zero8();

#pragma unroll 1
  for (int k0 = 0; k0 < DM; k0 += 32) {
    v16bf av[4], bv[4];
#pragma unroll
    for (int mi = 0; mi < 4; ++mi) {
      const __bf16* p = a0 + (size_t)(16 * mi) * DM + k0;
      av[mi] = ld_pair(p + 16 * sel, p + 8 + 16 * sel);
    }
#pragma unroll
    for (int ni = 0; ni < 4; ++ni)
      bv[ni] = ld_c16(b0 + (size_t)(16 * ni) * DM + k0 + 16 * sel);
#pragma unroll
    for (int mi = 0; mi < 4; ++mi)
#pragma unroll
      for (int ni = 0; ni < 4; ++ni)
        acc[mi][ni] = wmma_bf16(av[mi], bv[ni], acc[mi][ni]);
  }

  const int buf = e0 >> 10;           // 0=Q 1=K 2=V (uniform per wave)
  const int h = (e0 & 1023) >> 6;     // whole head per wave
  const int b = m0 >> 11;
  const int sq = m0 & 2047;
  if (buf < 2) {
    __bf16* dst = (buf == 0 ? Q : K) + (size_t)(b * NH + h) * SS * DH;
    const float scale = (buf == 0) ? 0.125f : 1.0f;
#pragma unroll
    for (int mi = 0; mi < 4; ++mi)
#pragma unroll
      for (int ni = 0; ni < 4; ++ni)
#pragma unroll
        for (int r = 0; r < 8; ++r)
          dst[(size_t)(sq + 16 * mi + r + 8 * sel) * DH + 16 * ni + lh] =
              f2bf(acc[mi][ni][r] * scale);
  } else {
    // C-layout rows (s) are contiguous in Vt -> packed 16B stores
#pragma unroll
    for (int ni = 0; ni < 4; ++ni) {
      __bf16* vt = Vt + ((size_t)(b * NH + h) * DH + 16 * ni + lh) * SS + sq;
#pragma unroll
      for (int mi = 0; mi < 4; ++mi) {
        v8bf pk;
#pragma unroll
        for (int r = 0; r < 8; ++r) pk[r] = f2bf(acc[mi][ni][r]);
        *(v8bf*)(vt + 16 * mi + 8 * sel) = pk;
      }
    }
  }
}

// ---------------------------------------------------------------- flash attn
// One wave per (b,h,16-query tile); 32-key blocks; online softmax.
__global__ void flash_attn(const __bf16* __restrict__ Q, const __bf16* __restrict__ K,
                           const __bf16* __restrict__ Vt, __bf16* __restrict__ O) {
  __shared__ __attribute__((aligned(16))) __bf16 pbuf[8 * 16 * 40];
  const int wave = threadIdx.x >> 5, lane = threadIdx.x & 31;
  const int lh = lane & 15, sel = lane >> 4;
  __bf16* lw = pbuf + wave * 16 * 40;  // row stride 40 elems (80B, 16B aligned)
  const int task = blockIdx.x * 8 + wave;  // BB*NH*(SS/16) = 4096
  const int qt = task & 127;
  const int bh = task >> 7;
  const int m0 = qt * 16;
  const __bf16* qrow = Q + ((size_t)bh * SS + m0 + lh) * DH;
  const __bf16* kbase = K + (size_t)bh * SS * DH;
  const __bf16* vbase = Vt + (size_t)bh * DH * SS;

  // Q A-fragments for dh chunks [0,32) and [32,64), reused every key block
  v16bf aq0 = ld_pair(qrow + 16 * sel, qrow + 8 + 16 * sel);
  v16bf aq1 = ld_pair(qrow + 32 + 16 * sel, qrow + 40 + 16 * sel);

  v8f o0 = zero8(), o1 = zero8(), o2 = zero8(), o3 = zero8();
  float mrow[8], lrow[8];
#pragma unroll
  for (int r = 0; r < 8; ++r) { mrow[r] = -1e30f; lrow[r] = 0.f; }

  const int nb = (m0 + 16 + 31) >> 5;
  for (int ib = 0; ib < nb; ++ib) {
    const int kb = ib * 32;
    // ---- scores: 16 queries x 32 keys (two 16-wide tiles, K=64 over dh)
    v8f s0 = zero8(), s1 = zero8();
    {
      const __bf16* k0p = kbase + (size_t)(kb + lh) * DH;
      const __bf16* k1p = kbase + (size_t)(kb + 16 + lh) * DH;
      s0 = wmma_bf16(aq0, ld_c16(k0p + 16 * sel), s0);
      s0 = wmma_bf16(aq1, ld_c16(k0p + 32 + 16 * sel), s0);
      s1 = wmma_bf16(aq0, ld_c16(k1p + 16 * sel), s1);
      s1 = wmma_bf16(aq1, ld_c16(k1p + 32 + 16 * sel), s1);
    }
    // ---- online softmax (rows = queries; keys across lanes within half)
    const int kA = kb + lh, kB = kb + 16 + lh;
#pragma unroll
    for (int r = 0; r < 8; ++r) {
      const int m = m0 + r + 8 * sel;
      float v0 = (kA <= m) ? s0[r] : -1e30f;   // causal mask
      float v1 = (kB <= m) ? s1[r] : -1e30f;
      float bm = rmax16(fmaxf(v0, v1));
      float mn = fmaxf(mrow[r], bm);
      float c = __expf(mrow[r] - mn);
      float p0 = __expf(v0 - mn);
      float p1 = __expf(v1 - mn);
      lrow[r] = lrow[r] * c + rsum16(p0 + p1);
      mrow[r] = mn;
      o0[r] *= c; o1[r] *= c; o2[r] *= c; o3[r] *= c;
      lw[(r + 8 * sel) * 40 + lh] = f2bf(p0);        // P[row][key]
      lw[(r + 8 * sel) * 40 + 16 + lh] = f2bf(p1);
    }
    // ---- C-layout -> A-layout transpose through per-wave LDS
    asm volatile("s_wait_dscnt 0" ::: "memory");  // LDS is in-order per wave
    v16bf ap = ld_pair(lw + lh * 40 + 16 * sel, lw + lh * 40 + 8 + 16 * sel);
    asm volatile("" ::: "memory");
    // ---- O += P @ V  (B fragments contiguous thanks to Vt layout)
    o0 = wmma_bf16(ap, ld_c16(vbase + (size_t)(0 + lh) * SS + kb + 16 * sel), o0);
    o1 = wmma_bf16(ap, ld_c16(vbase + (size_t)(16 + lh) * SS + kb + 16 * sel), o1);
    o2 = wmma_bf16(ap, ld_c16(vbase + (size_t)(32 + lh) * SS + kb + 16 * sel), o2);
    o3 = wmma_bf16(ap, ld_c16(vbase + (size_t)(48 + lh) * SS + kb + 16 * sel), o3);
  }
  // ---- normalize + store heads re-interleaved as [B,S,D] bf16
  const int b = bh >> 4, h = bh & 15;
  __bf16* ob = O + ((size_t)(b * SS + m0)) * DM + h * DH;
#pragma unroll
  for (int r = 0; r < 8; ++r) {
    float inv = 1.f / lrow[r];
    size_t row = (size_t)(r + 8 * sel) * DM;
    ob[row + 0 + lh]  = f2bf(o0[r] * inv);
    ob[row + 16 + lh] = f2bf(o1[r] * inv);
    ob[row + 32 + lh] = f2bf(o2[r] * inv);
    ob[row + 48 + lh] = f2bf(o3[r] * inv);
  }
}

// ---------------------------------------------------------------- out proj
// Same 64x64 register-blocked structure; f32 epilogue straight to d_out.
__global__ __launch_bounds__(256, 1) void out_proj(
    const __bf16* __restrict__ A, const __bf16* __restrict__ W,
    float* __restrict__ out) {
  const int wave = threadIdx.x >> 5, lane = threadIdx.x & 31;
  const int lh = lane & 15, sel = lane >> 4;
  const int NT = DM / 64;  // 16 column blocks
  const int tile = blockIdx.x * 8 + wave;
  const int m0 = (tile / NT) * 64;
  const int e0 = (tile % NT) * 64;
  const __bf16* a0 = A + (size_t)(m0 + lh) * DM;
  const __bf16* b0 = W + (size_t)(e0 + lh) * DM;

  v8f acc[4][4];
#pragma unroll
  for (int mi = 0; mi < 4; ++mi)
#pragma unroll
    for (int ni = 0; ni < 4; ++ni) acc[mi][ni] = zero8();

#pragma unroll 1
  for (int k0 = 0; k0 < DM; k0 += 32) {
    v16bf av[4], bv[4];
#pragma unroll
    for (int mi = 0; mi < 4; ++mi) {
      const __bf16* p = a0 + (size_t)(16 * mi) * DM + k0;
      av[mi] = ld_pair(p + 16 * sel, p + 8 + 16 * sel);
    }
#pragma unroll
    for (int ni = 0; ni < 4; ++ni)
      bv[ni] = ld_c16(b0 + (size_t)(16 * ni) * DM + k0 + 16 * sel);
#pragma unroll
    for (int mi = 0; mi < 4; ++mi)
#pragma unroll
      for (int ni = 0; ni < 4; ++ni)
        acc[mi][ni] = wmma_bf16(av[mi], bv[ni], acc[mi][ni]);
  }

#pragma unroll
  for (int mi = 0; mi < 4; ++mi)
#pragma unroll
    for (int ni = 0; ni < 4; ++ni)
#pragma unroll
      for (int r = 0; r < 8; ++r)
        out[(size_t)(m0 + 16 * mi + r + 8 * sel) * DM + e0 + 16 * ni + lh] =
            acc[mi][ni][r];
}

// ---------------------------------------------------------------- launcher
extern "C" void kernel_launch(void* const* d_in, const int* in_sizes, int n_in,
                              void* d_out, int out_size, void* d_ws, size_t ws_size,
                              hipStream_t stream) {
  (void)in_sizes; (void)n_in; (void)out_size; (void)ws_size;
  const float* x     = (const float*)d_in[0];
  // d_in[1] = causal mask, known analytically -> unused
  const float* w_qkv = (const float*)d_in[2];
  const float* w_o   = (const float*)d_in[3];

  uint8_t* ws = (uint8_t*)d_ws;
  size_t off = 0;
  const size_t SZ_XD = (size_t)BB * SS * DM * 2;  // 8 MB each
  __bf16* Xbf  = (__bf16*)(ws + off); off += SZ_XD;
  __bf16* Wqkv = (__bf16*)(ws + off); off += (size_t)3 * DM * DM * 2;
  __bf16* Wo   = (__bf16*)(ws + off); off += (size_t)DM * DM * 2;
  __bf16* Q    = (__bf16*)(ws + off); off += SZ_XD;
  __bf16* K    = (__bf16*)(ws + off); off += SZ_XD;
  __bf16* Vt   = (__bf16*)(ws + off); off += SZ_XD;
  __bf16* O    = (__bf16*)(ws + off); off += SZ_XD;  // total 48 MB

  f32_to_bf16<<<2048, 256, 0, stream>>>(x, Xbf, BB * SS * DM);
  f32_to_bf16<<<2048, 256, 0, stream>>>(w_qkv, Wqkv, 3 * DM * DM);
  f32_to_bf16<<<1024, 256, 0, stream>>>(w_o, Wo, DM * DM);

  // (4096/64)*(3072/64) = 3072 wave tiles, 8 waves/block
  qkv_gemm<<<3072 / 8, 256, 0, stream>>>(Xbf, Wqkv, Q, K, Vt);
  flash_attn<<<4096 / 8, 256, 0, stream>>>(Q, K, Vt, O);
  // (4096/64)*(1024/64) = 1024 wave tiles
  out_proj<<<1024 / 8, 256, 0, stream>>>(O, Wo, (float*)d_out);
}
